// BipartiteEncoder_85298050498706
// MI455X (gfx1250) — compile-verified, matrix operand order checked by reading, not run on hardware
//
#include <hip/hip_runtime.h>
#include <hip/hip_bf16.h>

// ---------------------------------------------------------------------------
// BipartiteEncoder for MI455X (gfx1250, wave32, WMMA).
// Dead cross-attention branch skipped; embed+pool branches fused; hyperedge
// residual MLP runs on v_wmma_f32_16x16x32_f16 with fp32 accumulate.
// ---------------------------------------------------------------------------

typedef __attribute__((ext_vector_type(16))) _Float16 v16h;
typedef __attribute__((ext_vector_type(8)))  _Float16 v8h;
typedef __attribute__((ext_vector_type(8)))  float    v8f;

union Frag16 { v16h v; v8h h[2]; };

#define NB   2048   // graphs
#define PPG  128    // particles per graph
#define EPG  256    // edges per graph
#define HPG  32     // hyperedges per graph
#define PH   256
#define EH   128
#define HH   128
#define LD   64
#define NH_ROWS (NB * HPG)   // 65536 hyperedge rows

__device__ __forceinline__ float gelu_exact(float x) {
    return 0.5f * x * (1.0f + erff(x * 0.7071067811865475f));
}
__device__ __forceinline__ float wsum32(float v) {
    #pragma unroll
    for (int m = 16; m >= 1; m >>= 1) v += __shfl_xor(v, m, 32);
    return v;
}
__device__ __forceinline__ float wsum16(float v) {   // reduce within 16-lane groups
    #pragma unroll
    for (int m = 8; m >= 1; m >>= 1) v += __shfl_xor(v, m, 32);
    return v;
}

// --------------------------- weight convert --------------------------------
// W1 (128x256) -> W1T f16 [n=256][k=128]; W2 (256x128) -> W2T f16 [n=128][k=256]
__global__ void k_convert_w(const float* __restrict__ l1w0, const float* __restrict__ l2w0,
                            const float* __restrict__ l1w1, const float* __restrict__ l2w1,
                            _Float16* __restrict__ w1t0, _Float16* __restrict__ w2t0,
                            _Float16* __restrict__ w1t1, _Float16* __restrict__ w2t1) {
    int idx = blockIdx.x * 256 + threadIdx.x;         // 4 * 32768 total
    int region = idx >> 15;
    int r = idx & 32767;
    const float* src; _Float16* dst;
    int n, k, din;
    if ((region & 1) == 0) { // W1: n<256, k<128, src row-major [k][256]
        n = r >> 7; k = r & 127; din = 256;
        src = (region == 0) ? l1w0 : l1w1;
        dst = (region == 0) ? w1t0 : w1t1;
        dst[n * 128 + k] = (_Float16)src[k * din + n];
    } else {                 // W2: n<128, k<256, src row-major [k][128]
        n = r >> 8; k = r & 255; din = 128;
        src = (region == 1) ? l2w0 : l2w1;
        dst = (region == 1) ? w2t0 : w2t1;
        dst[n * 256 + k] = (_Float16)src[k * din + n];
    }
}

// ----------------- particle branch: embed + LN + gelu + pool ---------------
__global__ void k_particle_pool(const float* __restrict__ x, const float* __restrict__ w,
                                const float* __restrict__ b, const float* __restrict__ lg,
                                const float* __restrict__ lb, float* __restrict__ pooled) {
    __shared__ float acc[PH];
    const int g = blockIdx.x, tid = threadIdx.x;
    const int lane = tid & 31, wv = tid >> 5;
    acc[tid] = 0.0f;
    __syncthreads();
    float lsum[8] = {0, 0, 0, 0, 0, 0, 0, 0};
    for (int r = wv; r < PPG; r += 8) {
        const float* xr = x + (size_t)(g * PPG + r) * 4;
        float x0 = xr[0], x1 = xr[1], x2 = xr[2], x3 = xr[3];
        float y[8]; float ps = 0.0f;
        #pragma unroll
        for (int i = 0; i < 8; ++i) {
            int f = lane * 8 + i;
            float v = b[f] + x0 * w[0 * PH + f] + x1 * w[1 * PH + f]
                           + x2 * w[2 * PH + f] + x3 * w[3 * PH + f];
            y[i] = v; ps += v;
        }
        float mean = wsum32(ps) * (1.0f / PH);
        float vs = 0.0f;
        #pragma unroll
        for (int i = 0; i < 8; ++i) { float d = y[i] - mean; vs += d * d; }
        float rstd = rsqrtf(wsum32(vs) * (1.0f / PH) + 1e-5f);
        #pragma unroll
        for (int i = 0; i < 8; ++i) {
            int f = lane * 8 + i;
            lsum[i] += gelu_exact((y[i] - mean) * rstd * lg[f] + lb[f]);
        }
    }
    #pragma unroll
    for (int i = 0; i < 8; ++i) atomicAdd(&acc[lane * 8 + i], lsum[i]);
    __syncthreads();
    pooled[(size_t)g * PH + tid] = acc[tid] * (1.0f / PPG);
}

// ------------------- edge branch: embed + LN + gelu + pool -----------------
__global__ void k_edge_pool(const float* __restrict__ x, const float* __restrict__ w,
                            const float* __restrict__ b, const float* __restrict__ lg,
                            const float* __restrict__ lb, float* __restrict__ pooled) {
    __shared__ float acc[EH];
    const int g = blockIdx.x, tid = threadIdx.x;
    const int lane = tid & 31, wv = tid >> 5;
    if (tid < EH) acc[tid] = 0.0f;
    __syncthreads();
    float lsum[4] = {0, 0, 0, 0};
    for (int r = wv; r < EPG; r += 8) {
        const float* xr = x + (size_t)(g * EPG + r) * 5;
        float x0 = xr[0], x1 = xr[1], x2 = xr[2], x3 = xr[3], x4 = xr[4];
        float y[4]; float ps = 0.0f;
        #pragma unroll
        for (int i = 0; i < 4; ++i) {
            int f = lane * 4 + i;
            float v = b[f] + x0 * w[0 * EH + f] + x1 * w[1 * EH + f] + x2 * w[2 * EH + f]
                           + x3 * w[3 * EH + f] + x4 * w[4 * EH + f];
            y[i] = v; ps += v;
        }
        float mean = wsum32(ps) * (1.0f / EH);
        float vs = 0.0f;
        #pragma unroll
        for (int i = 0; i < 4; ++i) { float d = y[i] - mean; vs += d * d; }
        float rstd = rsqrtf(wsum32(vs) * (1.0f / EH) + 1e-5f);
        #pragma unroll
        for (int i = 0; i < 4; ++i) {
            int f = lane * 4 + i;
            lsum[i] += gelu_exact((y[i] - mean) * rstd * lg[f] + lb[f]);
        }
    }
    #pragma unroll
    for (int i = 0; i < 4; ++i) atomicAdd(&acc[lane * 4 + i], lsum[i]);
    __syncthreads();
    if (tid < EH) pooled[(size_t)g * EH + tid] = acc[tid] * (1.0f / EPG);
}

// ---------------------- hyperedge embed (fp32 -> ws) -----------------------
__global__ void k_he_embed(const float* __restrict__ x, const float* __restrict__ w,
                           const float* __restrict__ b, const float* __restrict__ lg,
                           const float* __restrict__ lb, float* __restrict__ hx) {
    const int tid = threadIdx.x, lane = tid & 31, wv = tid >> 5;
    const long row = (long)blockIdx.x * 8 + wv;
    const float* xr = x + row * 2;
    float x0 = xr[0], x1 = xr[1];
    float y[4]; float ps = 0.0f;
    #pragma unroll
    for (int i = 0; i < 4; ++i) {
        int f = lane * 4 + i;
        float v = b[f] + x0 * w[0 * HH + f] + x1 * w[1 * HH + f];
        y[i] = v; ps += v;
    }
    float mean = wsum32(ps) * (1.0f / HH);
    float vs = 0.0f;
    #pragma unroll
    for (int i = 0; i < 4; ++i) { float d = y[i] - mean; vs += d * d; }
    float rstd = rsqrtf(wsum32(vs) * (1.0f / HH) + 1e-5f);
    #pragma unroll
    for (int i = 0; i < 4; ++i) {
        int f = lane * 4 + i;
        hx[row * HH + f] = gelu_exact((y[i] - mean) * rstd * lg[f] + lb[f]);
    }
}

// ------------- hyperedge residual MLP block (WMMA f16 -> f32) --------------
// hx <- hx + W2( gelu(LN( W1(hx)+b1 )) ) + b2 ; one 16-row M-tile per block.
__global__ void k_he_mlp(float* __restrict__ hx,
                         const _Float16* __restrict__ w1t, const float* __restrict__ b1,
                         const float* __restrict__ lng, const float* __restrict__ lnb,
                         const _Float16* __restrict__ w2t, const float* __restrict__ b2) {
    __shared__ __attribute__((aligned(16))) float    sx[16][HH];        // 8 KB residual
    __shared__ __attribute__((aligned(16))) _Float16 sa[16][HH];        // 4 KB A (f16)
    __shared__ __attribute__((aligned(16))) float    st[16][2 * HH];    // 16 KB GEMM1 out
    __shared__ __attribute__((aligned(16))) _Float16 sth[16][2 * HH];   // 8 KB gelu(LN) f16
    const int tid = threadIdx.x, lane = tid & 31, wv = tid >> 5;
    const int half = lane >> 4, l16 = lane & 15;
    const long row0 = (long)blockIdx.x * 16;

    // stage 16x128 activations
    for (int i = tid; i < 16 * HH; i += 256) {
        int r = i >> 7, k = i & 127;
        float v = hx[(row0 + r) * HH + k];
        sx[r][k] = v;
        sa[r][k] = (_Float16)v;
    }
    __syncthreads();

    // preload the 4 A fragments (K = 128 = 4 x 32), shared across N-chunks
    Frag16 afr[4];
    #pragma unroll
    for (int kc = 0; kc < 4; ++kc) {
        afr[kc].h[0] = *(const v8h*)&sa[l16][kc * 32 + 8 * half];
        afr[kc].h[1] = *(const v8h*)&sa[l16][kc * 32 + 16 + 8 * half];
    }

    // GEMM1: N = 256 -> 16 chunks, wave wv owns chunks {2wv, 2wv+1}
    #pragma unroll
    for (int c = 0; c < 2; ++c) {
        const int n0 = (wv * 2 + c) * 16;
        v8f acc = {};
        #pragma unroll
        for (int kc = 0; kc < 4; ++kc) {
            Frag16 bf;
            const _Float16* bp = w1t + (n0 + l16) * HH + kc * 32 + 8 * half;
            bf.h[0] = *(const v8h*)bp;
            bf.h[1] = *(const v8h*)(bp + 16);
            acc = __builtin_amdgcn_wmma_f32_16x16x32_f16(false, afr[kc].v, false, bf.v,
                                                         (short)0, acc, false, false);
        }
        const int col = n0 + l16;
        const float bb = b1[col];
        #pragma unroll
        for (int j = 0; j < 8; ++j) st[j + 8 * half][col] = acc[j] + bb;
    }
    __syncthreads();

    // LN(256) + gelu; 16 threads per row, 16 cols per thread
    {
        const int r = tid >> 4, sub = tid & 15;
        float vals[16]; float ps = 0.0f;
        #pragma unroll
        for (int i = 0; i < 16; ++i) { vals[i] = st[r][sub + 16 * i]; ps += vals[i]; }
        float mean = wsum16(ps) * (1.0f / (2 * HH));
        float vs = 0.0f;
        #pragma unroll
        for (int i = 0; i < 16; ++i) { float d = vals[i] - mean; vs += d * d; }
        float rstd = rsqrtf(wsum16(vs) * (1.0f / (2 * HH)) + 1e-5f);
        #pragma unroll
        for (int i = 0; i < 16; ++i) {
            int cidx = sub + 16 * i;
            float t = (vals[i] - mean) * rstd * lng[cidx] + lnb[cidx];
            sth[r][cidx] = (_Float16)gelu_exact(t);
        }
    }
    __syncthreads();

    // GEMM2: K = 256 (8 chunks), N = 128 -> wave wv owns chunk wv
    {
        const int n0 = wv * 16;
        v8f acc = {};
        #pragma unroll
        for (int kc = 0; kc < 8; ++kc) {
            Frag16 af, bf;
            af.h[0] = *(const v8h*)&sth[l16][kc * 32 + 8 * half];
            af.h[1] = *(const v8h*)&sth[l16][kc * 32 + 16 + 8 * half];
            const _Float16* bp = w2t + (n0 + l16) * (2 * HH) + kc * 32 + 8 * half;
            bf.h[0] = *(const v8h*)bp;
            bf.h[1] = *(const v8h*)(bp + 16);
            acc = __builtin_amdgcn_wmma_f32_16x16x32_f16(false, af.v, false, bf.v,
                                                         (short)0, acc, false, false);
        }
        const int col = n0 + l16;
        const float bb = b2[col];
        #pragma unroll
        for (int j = 0; j < 8; ++j) {
            int r = j + 8 * half;
            hx[(row0 + r) * HH + col] = acc[j] + bb + sx[r][col];
        }
    }
}

// --------------------------- hyperedge pool --------------------------------
__global__ void k_he_pool(const float* __restrict__ hx, float* __restrict__ pooled) {
    const int g = blockIdx.x, f = threadIdx.x;
    float s = 0.0f;
    #pragma unroll 8
    for (int r = 0; r < HPG; ++r) s += hx[((long)g * HPG + r) * HH + f];
    pooled[(size_t)g * HH + f] = s * (1.0f / HPG);
}

// ------------------------------ fusion head --------------------------------
__global__ void k_fusion(const float* __restrict__ pp, const float* __restrict__ hp,
                         const float* __restrict__ ep,
                         const float* __restrict__ f1w, const float* __restrict__ f1b,
                         const float* __restrict__ lng, const float* __restrict__ lnb,
                         const float* __restrict__ f2w, const float* __restrict__ f2b,
                         const float* __restrict__ muw, const float* __restrict__ mub,
                         const float* __restrict__ lvw, const float* __restrict__ lvb,
                         float* __restrict__ out) {
    __shared__ float sf[8][PH + HH + EH];   // 512 fused feats per graph
    __shared__ float s1[8][2 * LD];
    __shared__ float s2[8][2 * LD];
    const int tid = threadIdx.x, lane = tid & 31, wv = tid >> 5;
    const int g = blockIdx.x * 8 + wv;
    for (int i = lane; i < 512; i += 32) {
        float v;
        if (i < 256)      v = pp[(size_t)g * PH + i];
        else if (i < 384) v = hp[(size_t)g * HH + (i - 256)];
        else              v = ep[(size_t)g * EH + (i - 384)];
        sf[wv][i] = v;
    }
    __syncthreads();
    // f1: 512 -> 128, lane owns 4 cols
    float y[4];
    #pragma unroll
    for (int c = 0; c < 4; ++c) {
        const int col = lane * 4 + c;
        float a = f1b[col];
        #pragma unroll 8
        for (int i = 0; i < 512; ++i) a += sf[wv][i] * f1w[i * 128 + col];
        y[c] = a;
    }
    float ps = y[0] + y[1] + y[2] + y[3];
    float mean = wsum32(ps) * (1.0f / 128.0f);
    float vs = 0.0f;
    #pragma unroll
    for (int c = 0; c < 4; ++c) { float d = y[c] - mean; vs += d * d; }
    float rstd = rsqrtf(wsum32(vs) * (1.0f / 128.0f) + 1e-5f);
    #pragma unroll
    for (int c = 0; c < 4; ++c) {
        const int col = lane * 4 + c;
        s1[wv][col] = gelu_exact((y[c] - mean) * rstd * lng[col] + lnb[col]);
    }
    __syncthreads();
    // f2: 128 -> 128
    #pragma unroll
    for (int c = 0; c < 4; ++c) {
        const int col = lane * 4 + c;
        float a = f2b[col];
        #pragma unroll 8
        for (int i = 0; i < 128; ++i) a += s1[wv][i] * f2w[i * 128 + col];
        s2[wv][col] = a;
    }
    __syncthreads();
    // mu / logvar: 128 -> 64 each; lane owns 2 cols
    #pragma unroll
    for (int c = 0; c < 2; ++c) {
        const int col = lane * 2 + c;
        float am = mub[col], al = lvb[col];
        #pragma unroll 8
        for (int i = 0; i < 128; ++i) {
            float v = s2[wv][i];
            am += v * muw[i * 64 + col];
            al += v * lvw[i * 64 + col];
        }
        out[(size_t)g * 64 + col] = am;
        out[(size_t)NB * 64 + (size_t)g * 64 + col] = al;
    }
}

// ---------------------------------------------------------------------------
extern "C" void kernel_launch(void* const* d_in, const int* in_sizes, int n_in,
                              void* d_out, int out_size, void* d_ws, size_t ws_size,
                              hipStream_t stream) {
    const float* particle_x  = (const float*)d_in[0];
    const float* hyperedge_x = (const float*)d_in[1];
    const float* edge_attr   = (const float*)d_in[2];
    // d_in[3] edge_index, d_in[4] n_particles, d_in[5] n_hyperedges:
    // segmentation is block-contiguous by construction -> not needed.

    // params flattened depth-first in dict order
    const float* pew   = (const float*)d_in[6];
    const float* peb   = (const float*)d_in[7];
    const float* pelng = (const float*)d_in[8];
    const float* pelnb = (const float*)d_in[9];
    const float* eew   = (const float*)d_in[10];
    const float* eeb   = (const float*)d_in[11];
    const float* eelng = (const float*)d_in[12];
    const float* eelnb = (const float*)d_in[13];
    const float* hew   = (const float*)d_in[14];
    const float* heb   = (const float*)d_in[15];
    const float* helng = (const float*)d_in[16];
    const float* helnb = (const float*)d_in[17];
    const float* l1w[2] = {(const float*)d_in[18], (const float*)d_in[24]};
    const float* l1b[2] = {(const float*)d_in[19], (const float*)d_in[25]};
    const float* lng[2] = {(const float*)d_in[20], (const float*)d_in[26]};
    const float* lnb[2] = {(const float*)d_in[21], (const float*)d_in[27]};
    const float* l2w[2] = {(const float*)d_in[22], (const float*)d_in[28]};
    const float* l2b[2] = {(const float*)d_in[23], (const float*)d_in[29]};
    // 30..37: q/k/v/o -> dead code (cross_features unused)
    const float* f1w  = (const float*)d_in[38];
    const float* f1b  = (const float*)d_in[39];
    const float* flng = (const float*)d_in[40];
    const float* flnb = (const float*)d_in[41];
    const float* f2w  = (const float*)d_in[42];
    const float* f2b  = (const float*)d_in[43];
    const float* muw  = (const float*)d_in[44];
    const float* mub  = (const float*)d_in[45];
    const float* lvw  = (const float*)d_in[46];
    const float* lvb  = (const float*)d_in[47];

    // workspace layout
    char* ws = (char*)d_ws;
    constexpr size_t HX_B = (size_t)NH_ROWS * HH * sizeof(float);   // 32 MB
    constexpr size_t PP_B = (size_t)NB * PH * sizeof(float);
    constexpr size_t EP_B = (size_t)NB * EH * sizeof(float);
    constexpr size_t HP_B = (size_t)NB * HH * sizeof(float);
    constexpr size_t WT_B = (size_t)32768 * sizeof(_Float16);
    float*    hx   = (float*)(ws);
    float*    pp   = (float*)(ws + HX_B);
    float*    ep   = (float*)(ws + HX_B + PP_B);
    float*    hp   = (float*)(ws + HX_B + PP_B + EP_B);
    _Float16* w1t0 = (_Float16*)(ws + HX_B + PP_B + EP_B + HP_B);
    _Float16* w2t0 = (_Float16*)(ws + HX_B + PP_B + EP_B + HP_B + WT_B);
    _Float16* w1t1 = (_Float16*)(ws + HX_B + PP_B + EP_B + HP_B + 2 * WT_B);
    _Float16* w2t1 = (_Float16*)(ws + HX_B + PP_B + EP_B + HP_B + 3 * WT_B);

    k_convert_w<<<512, 256, 0, stream>>>(l1w[0], l2w[0], l1w[1], l2w[1],
                                         w1t0, w2t0, w1t1, w2t1);
    k_particle_pool<<<NB, 256, 0, stream>>>(particle_x, pew, peb, pelng, pelnb, pp);
    k_edge_pool<<<NB, 256, 0, stream>>>(edge_attr, eew, eeb, eelng, eelnb, ep);
    k_he_embed<<<NH_ROWS / 8, 256, 0, stream>>>(hyperedge_x, hew, heb, helng, helnb, hx);
    k_he_mlp<<<NH_ROWS / 16, 256, 0, stream>>>(hx, w1t0, l1b[0], lng[0], lnb[0], w2t0, l2b[0]);
    k_he_mlp<<<NH_ROWS / 16, 256, 0, stream>>>(hx, w1t1, l1b[1], lng[1], lnb[1], w2t1, l2b[1]);
    k_he_pool<<<NB, HH, 0, stream>>>(hx, hp);
    k_fusion<<<NB / 8, 256, 0, stream>>>(pp, hp, ep, f1w, f1b, flng, flnb,
                                         f2w, f2b, muw, mub, lvw, lvb, (float*)d_out);
}
